// Gatev_4_35828617183614
// MI455X (gfx1250) — matrix operand order repaired and checked by reading, not compiled, not run on hardware
//
#include <hip/hip_runtime.h>
#include <math.h>

// ---- problem constants (from reference setup_inputs) ----
#define B_   4
#define C_   64
#define KD_  16
#define H_   128
#define W_   128
#define KTOT 576           // C_ * 9 reduction length; K-index = j*64 + c
#define NCHUNK 18          // KTOT / 32  (K per v_wmma_f32_16x16x32_f16)
#define KWS_OFF 32768      // byte offset of k buffer inside d_ws
#define CPAD 72            // padded channel stride in LDS tile (halfs)

typedef __attribute__((ext_vector_type(16))) _Float16 v16h;
typedef __attribute__((ext_vector_type(8)))  float    v8f;

// ------------------------------------------------------------------
// Kernel 0: pack Wk into WMMA A-fragments (f16), reduction ordering
// K = j*64 + c  (tap-major, channel-minor).  Layout: this kernel's own
// 16-bit 16x32 A-matrix, contiguous K order (matches B fragment):
//   lanes 0-15 : M=lane,    K_local {0..15} in v0-7
//   lanes16-31 : M=lane-16, K_local {16..31} in v0-7
// Stored [chunk][lane][16 halfs] so the GEMM loads one v16h per chunk.
// ------------------------------------------------------------------
__global__ void prep_afrag(const float* __restrict__ Wk,
                           _Float16* __restrict__ afrag) {
  int tid = blockIdx.x * blockDim.x + threadIdx.x;
  if (tid >= NCHUNK * 32) return;
  int chunk = tid >> 5;
  int lane  = tid & 31;
  int m     = lane & 15;            // KD row
  int hi    = (lane >> 4) * 16;     // lane-half K offset
  #pragma unroll
  for (int e = 0; e < 16; ++e) {
    int g  = e >> 1;
    int hb = e & 1;
    int kl = hi + 2 * g + hb;
    int K  = chunk * 32 + kl;       // 0..575
    int j  = K >> 6;                // tap index ky*3+kx
    int c  = K & 63;                // input channel
    afrag[(chunk * 32 + lane) * 16 + e] = (_Float16)Wk[(m * C_ + c) * 9 + j];
  }
}

// ------------------------------------------------------------------
// Kernel 1: K-conv (3x3, C=64 -> KD=16) as implicit GEMM on the
// matrix pipe.  One block = one (b, h, 64-wide strip).  x halo tile
// staged in LDS as f16 with channel-contiguous layout [ky][col][c]
// so each chunk's B-fragment is ONE aligned 32-byte LDS load.
// 18x v_wmma_f32_16x16x32_f16 per wave (fully unrolled), bias init,
// ReLU, pixel-major [pix][16] f32 store (2x float4 per lane).
// ------------------------------------------------------------------
__global__ __launch_bounds__(128)
void kconv_wmma(const float* __restrict__ x,
                const float* __restrict__ bk,
                const _Float16* __restrict__ afrag,
                float* __restrict__ kws) {
  __shared__ _Float16 xs[3][66][CPAD];     // [ky][64+2 cols][C pad 72], ~28.5 KB

  const int TW = W_ / 64;                  // strips per row = 2
  int blk = blockIdx.x;
  int b   = blk / (H_ * TW);
  int r   = blk % (H_ * TW);
  int h   = r / TW;
  int wstart = (r % TW) * 64;
  int tid = threadIdx.x;

  // ---- stage x halo tile into LDS, transposing to [ky][col][c] ----
  // iterate col-fastest so global reads are coalesced
  for (int s = tid; s < 3 * C_ * 66; s += 128) {
    int ky  = s / (C_ * 66);
    int rr  = s - ky * (C_ * 66);
    int c   = rr / 66;
    int col = rr - c * 66;
    int gy = h - 1 + ky;
    int gx = wstart - 1 + col;
    float v = 0.f;
    if (gy >= 0 && gy < H_ && gx >= 0 && gx < W_)
      v = x[((b * C_ + c) * H_ + gy) * W_ + gx];
    xs[ky][col][c] = (_Float16)v;
  }
  __syncthreads();

  int lane  = tid & 31;
  int wvid  = tid >> 5;                    // which 16-pixel subtile
  int n     = lane & 15;                   // pixel column within subtile
  int kbase = (lane >> 4) << 4;            // B-fragment K_local base (0 / 16)
  int colbase = wvid * 16 + n;             // LDS column of this pixel (+kx)
  int mofs  = (lane >> 4) * 8;             // C/D: lanes>=16 hold rows 8..15

  // C initialized with bias bk[m]
  v8f acc;
  #pragma unroll
  for (int rr = 0; rr < 8; ++rr) acc[rr] = bk[mofs + rr];

  const v16h* afp = (const v16h*)afrag;
  const _Float16* xflat = &xs[0][0][0];

  #pragma unroll
  for (int kc = 0; kc < NCHUNK; ++kc) {
    // tap index is compile-time per unrolled iteration
    const int j  = kc >> 1;
    const int ky = j / 3;
    const int kx = j - ky * 3;
    int cstart = ((kc & 1) << 5) + kbase;  // starting channel (mult of 16)
    // 16 consecutive channels -> one aligned 32-byte LDS load
    const v16h bf = *(const v16h*)(xflat + (ky * 66 + colbase + kx) * CPAD + cstart);
    v16h a = afp[kc * 32 + lane];          // two b128 global loads
    acc = __builtin_amdgcn_wmma_f32_16x16x32_f16(
        /*neg_a=*/false, a, /*neg_b=*/false, bf,
        /*c_mod=*/(short)0, acc, /*reuse_a=*/false, /*reuse_b=*/false);
  }

  #pragma unroll
  for (int rr = 0; rr < 8; ++rr) acc[rr] = fmaxf(acc[rr], 0.f);   // ReLU

  int pix = (b * H_ + h) * W_ + wstart + wvid * 16 + n;
  float* dst = kws + pix * 16 + mofs;      // contiguous per lane
  *(float4*)(dst)     = make_float4(acc[0], acc[1], acc[2], acc[3]);
  *(float4*)(dst + 4) = make_float4(acc[4], acc[5], acc[6], acc[7]);
}

// ------------------------------------------------------------------
// Kernel 2: fused q (depthwise 3x3 + ReLU) / v / softmax / reduce.
// One thread per output element; bandwidth-dominant pass
// (y, z, k read once; out written once).
// ------------------------------------------------------------------
__global__ __launch_bounds__(256)
void fused_attn(const float* __restrict__ y, const float* __restrict__ z,
                const float* __restrict__ kws,
                const float* __restrict__ Wq, const float* __restrict__ bq,
                const float* __restrict__ wv, const float* __restrict__ bv,
                float* __restrict__ out) {
  __shared__ float sWq[C_ * 9];
  __shared__ float sbq[C_];
  __shared__ float swv[KD_];
  __shared__ float sbv[KD_];
  int tid = threadIdx.x;
  for (int s = tid; s < C_ * 9; s += 256) sWq[s] = Wq[s];
  if (tid < C_)  sbq[tid] = bq[tid];
  if (tid < KD_) { swv[tid] = wv[tid]; sbv[tid] = bv[tid]; }
  __syncthreads();

  int idx = blockIdx.x * 256 + tid;
  int w = idx & (W_ - 1);
  int h = (idx >> 7) & (H_ - 1);
  int c = (idx >> 14) & (C_ - 1);
  int b = idx >> 20;

  // depthwise 3x3 q on y (SAME padding)
  const float* yb  = y + ((size_t)(b * C_ + c) * H_) * W_;
  const float* wqc = sWq + c * 9;
  float q = sbq[c];
  #pragma unroll
  for (int ky = 0; ky < 3; ++ky) {
    int gy = h + ky - 1;
    if (gy < 0 || gy >= H_) continue;
    #pragma unroll
    for (int kx = 0; kx < 3; ++kx) {
      int gx = w + kx - 1;
      if (gx < 0 || gx >= W_) continue;
      q = fmaf(yb[gy * W_ + gx], wqc[ky * 3 + kx], q);
    }
  }
  q = fmaxf(q, 0.f);

  // k row: 16 contiguous floats per pixel (4x b128)
  const float4* kp = (const float4*)(kws + ((size_t)((b * H_ + h) * W_ + w)) * 16);
  float kv[16];
  #pragma unroll
  for (int i = 0; i < 4; ++i) {
    float4 t = kp[i];
    kv[4 * i + 0] = t.x; kv[4 * i + 1] = t.y;
    kv[4 * i + 2] = t.z; kv[4 * i + 3] = t.w;
  }

  float zv = z[idx];
  float sc[16];
  float m = -3.402823466e38f;
  #pragma unroll
  for (int kd = 0; kd < 16; ++kd) {
    sc[kd] = q * kv[kd];
    m = fmaxf(m, sc[kd]);
  }
  float den = 0.f, num = 0.f;
  #pragma unroll
  for (int kd = 0; kd < 16; ++kd) {
    float e  = __expf(sc[kd] - m);                          // TRANS pipe
    float vv = fmaxf(fmaf(zv, swv[kd], sbv[kd]), 0.f);      // v = relu(z*wv+bv)
    den += e;
    num = fmaf(e, vv, num);
  }
  out[idx] = num / den;
}

// ------------------------------------------------------------------
extern "C" void kernel_launch(void* const* d_in, const int* in_sizes, int n_in,
                              void* d_out, int out_size, void* d_ws, size_t ws_size,
                              hipStream_t stream) {
  (void)in_sizes; (void)n_in; (void)out_size; (void)ws_size;
  const float* x  = (const float*)d_in[0];
  const float* y  = (const float*)d_in[1];
  const float* z  = (const float*)d_in[2];
  const float* Wq = (const float*)d_in[3];
  const float* bq = (const float*)d_in[4];
  const float* Wk = (const float*)d_in[5];
  const float* bk = (const float*)d_in[6];
  const float* wv = (const float*)d_in[7];
  const float* bv = (const float*)d_in[8];
  float* out = (float*)d_out;

  _Float16* afrag = (_Float16*)d_ws;                       // 18432 B
  float*    kws   = (float*)((char*)d_ws + KWS_OFF);       // 4 MB: [B*H*W][16]

  prep_afrag<<<(NCHUNK * 32 + 127) / 128, 128, 0, stream>>>(Wk, afrag);
  kconv_wmma<<<B_ * H_ * (W_ / 64), 128, 0, stream>>>(x, bk, afrag, kws);
  fused_attn<<<(B_ * C_ * H_ * W_) / 256, 256, 0, stream>>>(
      y, z, kws, Wq, bq, wv, bv, out);
}
